// MambaBlock_34248069218985
// MI455X (gfx1250) — matrix operand order repaired and checked by reading, not compiled, not run on hardware
//
#include <hip/hip_runtime.h>
#include <hip/hip_bf16.h>

// ---------------------------------------------------------------------------
// MambaBlock for MI455X (gfx1250, wave32, WMMA).
// B=8, T=4096, D_MODEL=128, D_INNER=STATE=256, K=4.
// Scan dominates (~1.1 TFLOP) -> resident-state bf16 WMMA recurrence:
//   - A (256x256 bf16) pinned in VGPRs as WMMA B-fragments (8 frags/wave)
//   - state (16x256) ping-pongs through LDS in bf16, f32 accumulation
//   - 1 barrier/step, no LDS atomics, 2 independent WMMA chains for ILP
// ---------------------------------------------------------------------------

typedef __bf16 bf16_t;
typedef __attribute__((ext_vector_type(16))) __bf16 v16bf;
typedef __attribute__((ext_vector_type(8)))  float  v8f;

union Frag16 { v16bf v; uint4 u[2]; };

#define BSZ   8
#define TLEN  4096
#define DMOD  128
#define DINN  256
#define ROWS  ((size_t)BSZ * TLEN)      // 32768

#define WMMA_BF16(A, B, C) \
  __builtin_amdgcn_wmma_f32_16x16x32_bf16(false, (A), false, (B), (short)0, (C), false, false)

// ---------------------------------------------------------------------------
// K0: precompute bf16 copies of operands; A = -exp(A_log)
// ---------------------------------------------------------------------------
__global__ void precvt(const float* __restrict__ W_in, const float* __restrict__ A_log,
                       const float* __restrict__ W_out, const float* __restrict__ x,
                       bf16_t* __restrict__ W_inbf, bf16_t* __restrict__ Abf,
                       bf16_t* __restrict__ Woutbf, bf16_t* __restrict__ xbf) {
  long i0 = (long)blockIdx.x * blockDim.x + threadIdx.x;
  long stride = (long)gridDim.x * blockDim.x;
  for (long i = i0; i < 512 * 128; i += stride) W_inbf[i] = (bf16_t)W_in[i];
  for (long i = i0; i < 256 * 256; i += stride) Abf[i]   = (bf16_t)(-__expf(A_log[i]));
  for (long i = i0; i < 128 * 256; i += stride) Woutbf[i] = (bf16_t)W_out[i];
  for (long i = i0; i < (long)ROWS * DMOD; i += stride) xbf[i] = (bf16_t)x[i];
}

// ---------------------------------------------------------------------------
// K1: in_proj GEMM  xp(32768 x 512) = xbf(32768 x 128) @ W_in^T + b_in
// One 16x16 output tile per wave, K=128 in 4 WMMA steps (2 ILP chains).
// ---------------------------------------------------------------------------
__global__ __launch_bounds__(256) void gemm_in(const bf16_t* __restrict__ xbf,
                                               const bf16_t* __restrict__ Wbf,
                                               const float* __restrict__ b_in,
                                               float* __restrict__ xp) {
  const int lane = threadIdx.x & 31;
  const int wv   = threadIdx.x >> 5;
  const int tile = blockIdx.x * 8 + wv;      // 65536 tiles = 2048 x 32
  const int tm = tile >> 5;
  const int tn = tile & 31;
  const int h = lane >> 4;
  const int r = lane & 15;
  const int m0 = tm * 16, n0 = tn * 16;
  const bf16_t* arow = xbf + (size_t)(m0 + r) * 128;
  const bf16_t* brow = Wbf + (size_t)(n0 + r) * 128;
  v8f c0 = {0.f,0.f,0.f,0.f,0.f,0.f,0.f,0.f};
  v8f c1 = {0.f,0.f,0.f,0.f,0.f,0.f,0.f,0.f};
  #pragma unroll
  for (int kt = 0; kt < 4; ++kt) {
    Frag16 a, b;
    a.u[0] = *(const uint4*)(arow + kt * 32 + h * 16);
    a.u[1] = *(const uint4*)(arow + kt * 32 + h * 16 + 8);
    const uint4* bp = (const uint4*)(brow + kt * 32 + h * 16);
    b.u[0] = bp[0]; b.u[1] = bp[1];
    if (kt & 1) c1 = WMMA_BF16(a.v, b.v, c1);
    else        c0 = WMMA_BF16(a.v, b.v, c0);
  }
  const int n = n0 + r;
  const float bias = b_in[n];
  #pragma unroll
  for (int v = 0; v < 8; ++v) {
    const int m = m0 + v + h * 8;
    xp[(size_t)m * 512 + n] = c0[v] + c1[v] + bias;
  }
}

// ---------------------------------------------------------------------------
// K2: depthwise conv1d (K=4, left pad 2), reads x_in = xp[:, 0:256]
// ---------------------------------------------------------------------------
__global__ void dwconv(const float* __restrict__ xp, const float* __restrict__ conv_w,
                       const float* __restrict__ conv_b, float* __restrict__ u) {
  const long total = (long)ROWS * DINN;
  long idx0 = (long)blockIdx.x * blockDim.x + threadIdx.x;
  long stride = (long)gridDim.x * blockDim.x;
  for (long idx = idx0; idx < total; idx += stride) {
    const int  i   = (int)(idx & 255);
    const long row = idx >> 8;               // b*T + t
    const long b   = row >> 12;
    const int  t   = (int)(row & 4095);
    float acc = conv_b[i];
    #pragma unroll
    for (int k = 0; k < 4; ++k) {
      const int tt = t - 2 + k;
      if (tt >= 0 && tt < TLEN)
        acc += conv_w[i * 4 + k] * xp[(((b << 12) + tt) * 512) + i];
    }
    u[idx] = acc;
  }
}

// ---------------------------------------------------------------------------
// K3: the scan. 128 workgroups x 512 threads (16 waves).
// Each WG owns 16 rows (one b, 16 channels); per step per wave:
// 8 x v_wmma_f32_16x16x32_bf16 in two independent accumulation chains.
// One barrier per step: y-flush of step t-1 and u-prefetch of step t+1 are
// overlapped with step t's WMMA chain (double-buffered u_s / ypart).
// ---------------------------------------------------------------------------
__global__ __launch_bounds__(512) void scan(const bf16_t* __restrict__ Abf,
                                            const float* __restrict__ u,
                                            float* __restrict__ ys) {
  __shared__ bf16_t st[2][16 * 256];      // state double buffer (bf16)
  __shared__ float  u_s[2][16];           // u_t staging, double buffered
  __shared__ float  ypart[2][16][16];     // [buf][wave][m] row-sum partials

  const int tid  = threadIdx.x;
  const int lane = tid & 31;
  const int wv   = tid >> 5;          // 0..15 : output column tile
  const int h    = lane >> 4;
  const int r    = lane & 15;
  const int blk  = blockIdx.x;        // 0..127
  const int b    = blk >> 4;
  const int i0   = (blk & 15) << 4;
  const size_t base = ((size_t)b * TLEN) * 256 + i0;   // u and ys share layout

  // Persistent B-fragments: Bop[k=s][n=d] = A[d][s]; lane holds column d,
  // contiguous 16 K-values per lane-half.
  Frag16 bfr[8];
  {
    const bf16_t* brow = Abf + (size_t)(wv * 16 + r) * 256;
    #pragma unroll
    for (int kt = 0; kt < 8; ++kt) {
      const uint4* bp = (const uint4*)(brow + kt * 32 + h * 16);
      bfr[kt].u[0] = bp[0]; bfr[kt].u[1] = bp[1];
    }
  }

  for (int idx = tid; idx < 16 * 256; idx += 512) st[0][idx] = (bf16_t)0.0f;
  if (tid < 16) u_s[0][tid] = u[base + tid];
  __syncthreads();

  int p = 0;
  for (int t = 0; t < TLEN; ++t) {
    const int q = t & 1;

    // new_state[m, d] = sum_s state[m,s] * A[d,s]  (d in this wave's tile)
    v8f c0 = {0.f,0.f,0.f,0.f,0.f,0.f,0.f,0.f};
    v8f c1 = {0.f,0.f,0.f,0.f,0.f,0.f,0.f,0.f};
    const bf16_t* srow = &st[p][r * 256];
    #pragma unroll
    for (int kt = 0; kt < 8; kt += 2) {
      Frag16 a0, a1;
      a0.u[0] = *(const uint4*)(srow + kt * 32 + h * 16);
      a0.u[1] = *(const uint4*)(srow + kt * 32 + h * 16 + 8);
      a1.u[0] = *(const uint4*)(srow + (kt + 1) * 32 + h * 16);
      a1.u[1] = *(const uint4*)(srow + (kt + 1) * 32 + h * 16 + 8);
      c0 = WMMA_BF16(a0.v, bfr[kt].v,     c0);
      c1 = WMMA_BF16(a1.v, bfr[kt + 1].v, c1);
    }

    // Overlapped with the WMMA chain: flush y of step t-1, prefetch u of t+1.
    if (tid < 16) {
      if (t > 0) {
        float acc = 0.0f;
        #pragma unroll
        for (int w = 0; w < 16; ++w) acc += ypart[q ^ 1][w][tid];
        ys[base + (size_t)(t - 1) * 256 + tid] = acc;
      }
      if (t + 1 < TLEN) u_s[q ^ 1][tid] = u[base + (size_t)(t + 1) * 256 + tid];
    }

    // merge chains, add u_t, write next state (bf16), emit row-sum partials
    bf16_t* dst = &st[p ^ 1][0];
    const int d = (wv << 4) + r;
    #pragma unroll
    for (int v = 0; v < 8; ++v) {
      const float val = c0[v] + c1[v] + u_s[q][v + h * 8];
      dst[(v + h * 8) * 256 + d] = (bf16_t)val;
      float s = val;
      s += __shfl_xor(s, 1, 32);
      s += __shfl_xor(s, 2, 32);
      s += __shfl_xor(s, 4, 32);
      s += __shfl_xor(s, 8, 32);
      if (r == 0) ypart[q][wv][v + h * 8] = s;   // lanes 0 and 16
    }
    __syncthreads();
    p ^= 1;
  }

  // flush final step's y
  if (tid < 16) {
    float acc = 0.0f;
    #pragma unroll
    for (int w = 0; w < 16; ++w) acc += ypart[(TLEN - 1) & 1][w][tid];
    ys[base + (size_t)(TLEN - 1) * 256 + tid] = acc;
  }
}

// ---------------------------------------------------------------------------
// K4: y = ys * silu(x_res); LayerNorm(256); emit bf16 for out_proj GEMM.
// ---------------------------------------------------------------------------
__global__ __launch_bounds__(256) void gatenorm(const float* __restrict__ ys,
                                                const float* __restrict__ xp,
                                                const float* __restrict__ gamma,
                                                const float* __restrict__ beta,
                                                bf16_t* __restrict__ ybf) {
  const int row = blockIdx.x;
  const int i = threadIdx.x;
  __shared__ float red[256];
  const float g = xp[(size_t)row * 512 + 256 + i];
  const float sg = g / (1.0f + __expf(-g));
  const float val = ys[(size_t)row * 256 + i] * sg;

  red[i] = val; __syncthreads();
  for (int off = 128; off > 0; off >>= 1) { if (i < off) red[i] += red[i + off]; __syncthreads(); }
  const float mu = red[0] * (1.0f / 256.0f);
  __syncthreads();
  const float dv = val - mu;
  red[i] = dv * dv; __syncthreads();
  for (int off = 128; off > 0; off >>= 1) { if (i < off) red[i] += red[i + off]; __syncthreads(); }
  const float var = red[0] * (1.0f / 256.0f);
  const float nv = dv * rsqrtf(var + 1e-5f) * gamma[i] + beta[i];
  ybf[(size_t)row * 256 + i] = (bf16_t)nv;
}

// ---------------------------------------------------------------------------
// K5: out_proj GEMM  out(32768 x 128) = ybf @ W_out^T + b_out + x
// ---------------------------------------------------------------------------
__global__ __launch_bounds__(256) void gemm_out(const bf16_t* __restrict__ ybf,
                                                const bf16_t* __restrict__ Wbf,
                                                const float* __restrict__ b_out,
                                                const float* __restrict__ x,
                                                float* __restrict__ out) {
  const int lane = threadIdx.x & 31;
  const int wv   = threadIdx.x >> 5;
  const int tile = blockIdx.x * 8 + wv;      // 16384 tiles = 2048 x 8
  const int tm = tile >> 3;
  const int tn = tile & 7;
  const int h = lane >> 4;
  const int r = lane & 15;
  const int m0 = tm * 16, n0 = tn * 16;
  const bf16_t* arow = ybf + (size_t)(m0 + r) * 256;
  const bf16_t* brow = Wbf + (size_t)(n0 + r) * 256;
  v8f c0 = {0.f,0.f,0.f,0.f,0.f,0.f,0.f,0.f};
  v8f c1 = {0.f,0.f,0.f,0.f,0.f,0.f,0.f,0.f};
  #pragma unroll
  for (int kt = 0; kt < 8; ++kt) {
    Frag16 a, b;
    a.u[0] = *(const uint4*)(arow + kt * 32 + h * 16);
    a.u[1] = *(const uint4*)(arow + kt * 32 + h * 16 + 8);
    const uint4* bp = (const uint4*)(brow + kt * 32 + h * 16);
    b.u[0] = bp[0]; b.u[1] = bp[1];
    if (kt & 1) c1 = WMMA_BF16(a.v, b.v, c1);
    else        c0 = WMMA_BF16(a.v, b.v, c0);
  }
  const int n = n0 + r;
  const float bias = b_out[n];
  #pragma unroll
  for (int v = 0; v < 8; ++v) {
    const int m = m0 + v + h * 8;
    out[(size_t)m * 128 + n] = c0[v] + c1[v] + bias + x[(size_t)m * 128 + n];
  }
}

// ---------------------------------------------------------------------------
extern "C" void kernel_launch(void* const* d_in, const int* in_sizes, int n_in,
                              void* d_out, int out_size, void* d_ws, size_t ws_size,
                              hipStream_t stream) {
  (void)in_sizes; (void)n_in; (void)out_size; (void)ws_size;
  const float* x      = (const float*)d_in[0];
  const float* W_in   = (const float*)d_in[1];
  const float* b_in   = (const float*)d_in[2];
  const float* conv_w = (const float*)d_in[3];
  const float* conv_b = (const float*)d_in[4];
  const float* A_log  = (const float*)d_in[5];
  const float* gamma  = (const float*)d_in[6];
  const float* beta   = (const float*)d_in[7];
  const float* W_out  = (const float*)d_in[8];
  const float* b_out  = (const float*)d_in[9];
  float* out = (float*)d_out;

  char* ws = (char*)d_ws;
  size_t off = 0;
  auto alloc = [&](size_t bytes) -> char* {
    char* p = ws + off;
    off = (off + bytes + 255) & ~(size_t)255;
    return p;
  };
  bf16_t* W_inbf  = (bf16_t*)alloc((size_t)512 * 128 * sizeof(bf16_t));
  bf16_t* Abf     = (bf16_t*)alloc((size_t)256 * 256 * sizeof(bf16_t));
  bf16_t* Woutbf  = (bf16_t*)alloc((size_t)128 * 256 * sizeof(bf16_t));
  bf16_t* xbf     = (bf16_t*)alloc(ROWS * DMOD * sizeof(bf16_t));
  float*  xp      = (float*) alloc(ROWS * 512 * sizeof(float));
  float*  u       = (float*) alloc(ROWS * DINN * sizeof(float));
  float*  ysbuf   = (float*) alloc(ROWS * DINN * sizeof(float));
  bf16_t* ybf     = (bf16_t*)alloc(ROWS * DINN * sizeof(bf16_t));

  precvt  <<<4096, 256, 0, stream>>>(W_in, A_log, W_out, x, W_inbf, Abf, Woutbf, xbf);
  gemm_in <<<8192, 256, 0, stream>>>(xbf, W_inbf, b_in, xp);
  dwconv  <<<8192, 256, 0, stream>>>(xp, conv_w, conv_b, u);
  scan    <<<128, 512, 0, stream>>>(Abf, u, ysbuf);
  gatenorm<<<(int)ROWS, 256, 0, stream>>>(ysbuf, xp, gamma, beta, ybf);
  gemm_out<<<2048, 256, 0, stream>>>(ybf, Woutbf, b_out, x, out);
}
